// AngleEncodingQuantumNet_25726854103522
// MI455X (gfx1250) — matrix operand order repaired and checked by reading, not compiled, not run on hardware
//
#include <hip/hip_runtime.h>
#include <hip/hip_bf16.h>
#include <math.h>

#define BATCH 2048
#define PI_F 3.14159265358979f
#define FPAD 800   // feats K padded to 25*32 (784 valid + 16 zero)

typedef __attribute__((ext_vector_type(16))) _Float16 v16h;
typedef __attribute__((ext_vector_type(8)))  _Float16 v8h;
typedef __attribute__((ext_vector_type(8)))  float    v8f;

// ---------------------------------------------------------------------------
// conv helpers (direct 3x3 SAME conv, recomputed in stats + apply passes)
// ---------------------------------------------------------------------------
__device__ __forceinline__ float conv1_val(const float* __restrict__ x,
                                           const float* __restrict__ w,
                                           const float* __restrict__ bias,
                                           int b, int c, int oy, int ox) {
    float v = bias[c];
    const float* xb = x + (size_t)b * 784;
    const float* wc = w + c * 9;
#pragma unroll
    for (int dy = 0; dy < 3; ++dy) {
        int iy = oy + dy - 1;
        if ((unsigned)iy < 28u) {
#pragma unroll
            for (int dx = 0; dx < 3; ++dx) {
                int ix = ox + dx - 1;
                if ((unsigned)ix < 28u) v += xb[iy * 28 + ix] * wc[dy * 3 + dx];
            }
        }
    }
    return v;
}

__device__ __forceinline__ float conv2_val(const float* __restrict__ p1,   // [B,16,14,14]
                                           const float* __restrict__ w,    // [32,16,3,3]
                                           const float* __restrict__ bias,
                                           int b, int c, int oy, int ox) {
    float v = bias[c];
#pragma unroll
    for (int ci = 0; ci < 16; ++ci) {
        const float* xb = p1 + (((size_t)b * 16 + ci) * 14) * 14;
        const float* wc = w + ((c * 16 + ci) * 9);
#pragma unroll
        for (int dy = 0; dy < 3; ++dy) {
            int iy = oy + dy - 1;
            if ((unsigned)iy < 14u) {
#pragma unroll
                for (int dx = 0; dx < 3; ++dx) {
                    int ix = ox + dx - 1;
                    if ((unsigned)ix < 14u) v += xb[iy * 14 + ix] * wc[dy * 3 + dx];
                }
            }
        }
    }
    return v;
}

// ---------------------------------------------------------------------------
// BN batch-stat partial sums (deterministic 2-stage reduction)
// ---------------------------------------------------------------------------
__global__ __launch_bounds__(256)
void conv1_stats_kernel(const float* __restrict__ x, const float* __restrict__ w,
                        const float* __restrict__ bias, float* __restrict__ partial) {
    const int CH = 128;
    int chan = blockIdx.x / CH;
    int chunk = blockIdx.x % CH;
    const int NE = BATCH * 784;
    const int PER = NE / CH;               // 12544
    int start = chunk * PER;
    float s = 0.f, ss = 0.f;
    for (int e = start + threadIdx.x; e < start + PER; e += 256) {
        int b = e / 784, r = e % 784;
        float v = conv1_val(x, w, bias, b, chan, r / 28, r % 28);
        s += v; ss += v * v;
    }
    __shared__ float rs[256], rss[256];
    rs[threadIdx.x] = s; rss[threadIdx.x] = ss;
    __syncthreads();
    for (int off = 128; off > 0; off >>= 1) {
        if (threadIdx.x < off) { rs[threadIdx.x] += rs[threadIdx.x + off];
                                 rss[threadIdx.x] += rss[threadIdx.x + off]; }
        __syncthreads();
    }
    if (threadIdx.x == 0) {
        partial[(size_t)blockIdx.x * 2 + 0] = rs[0];
        partial[(size_t)blockIdx.x * 2 + 1] = rss[0];
    }
}

__global__ __launch_bounds__(256)
void conv2_stats_kernel(const float* __restrict__ p1, const float* __restrict__ w,
                        const float* __restrict__ bias, float* __restrict__ partial) {
    const int CH = 128;
    int chan = blockIdx.x / CH;
    int chunk = blockIdx.x % CH;
    const int NE = BATCH * 196;
    const int PER = NE / CH;               // 3136
    int start = chunk * PER;
    float s = 0.f, ss = 0.f;
    for (int e = start + threadIdx.x; e < start + PER; e += 256) {
        int b = e / 196, r = e % 196;
        float v = conv2_val(p1, w, bias, b, chan, r / 14, r % 14);
        s += v; ss += v * v;
    }
    __shared__ float rs[256], rss[256];
    rs[threadIdx.x] = s; rss[threadIdx.x] = ss;
    __syncthreads();
    for (int off = 128; off > 0; off >>= 1) {
        if (threadIdx.x < off) { rs[threadIdx.x] += rs[threadIdx.x + off];
                                 rss[threadIdx.x] += rss[threadIdx.x + off]; }
        __syncthreads();
    }
    if (threadIdx.x == 0) {
        partial[(size_t)blockIdx.x * 2 + 0] = rs[0];
        partial[(size_t)blockIdx.x * 2 + 1] = rss[0];
    }
}

__global__ __launch_bounds__(128)
void bn_finalize_kernel(const float* __restrict__ partial, const float* __restrict__ g,
                        const float* __restrict__ bb, float* __restrict__ scale,
                        float* __restrict__ shift, float invNE) {
    int c = blockIdx.x, tid = threadIdx.x;
    __shared__ float rs[128], rss[128];
    rs[tid]  = partial[((size_t)c * 128 + tid) * 2 + 0];
    rss[tid] = partial[((size_t)c * 128 + tid) * 2 + 1];
    __syncthreads();
    for (int off = 64; off > 0; off >>= 1) {
        if (tid < off) { rs[tid] += rs[tid + off]; rss[tid] += rss[tid + off]; }
        __syncthreads();
    }
    if (tid == 0) {
        float m = rs[0] * invNE;
        float var = rss[0] * invNE - m * m;
        float sc = g[c] * rsqrtf(var + 1e-5f);
        scale[c] = sc;
        shift[c] = bb[c] - m * sc;
    }
}

// ---------------------------------------------------------------------------
// conv(recompute) + bn + relu + 2x2 maxpool fused apply passes
// ---------------------------------------------------------------------------
__global__ __launch_bounds__(256)
void pool1_kernel(const float* __restrict__ x, const float* __restrict__ w,
                  const float* __restrict__ bias, const float* __restrict__ scale,
                  const float* __restrict__ shift, float* __restrict__ out) {
    int idx = blockIdx.x * 256 + threadIdx.x;       // [B,16,14,14]
    if (idx >= BATCH * 16 * 14 * 14) return;
    int ox = idx % 14, oy = (idx / 14) % 14, c = (idx / 196) % 16, b = idx / (196 * 16);
    float sc = scale[c], sh = shift[c];
    float m = -1e30f;
#pragma unroll
    for (int py = 0; py < 2; ++py)
#pragma unroll
        for (int px = 0; px < 2; ++px) {
            float v = conv1_val(x, w, bias, b, c, oy * 2 + py, ox * 2 + px);
            v = fmaxf(v * sc + sh, 0.f);
            m = fmaxf(m, v);
        }
    out[idx] = m;
}

__global__ __launch_bounds__(256)
void pool2_kernel(const float* __restrict__ p1, const float* __restrict__ w,
                  const float* __restrict__ bias, const float* __restrict__ scale,
                  const float* __restrict__ shift, float* __restrict__ h) {
    int idx = blockIdx.x * 256 + threadIdx.x;       // [B,32,7,7]
    if (idx >= BATCH * 32 * 7 * 7) return;
    int ox = idx % 7, oy = (idx / 7) % 7, c = (idx / 49) % 32, b = idx / (49 * 32);
    float sc = scale[c], sh = shift[c];
    float m = -1e30f;
#pragma unroll
    for (int py = 0; py < 2; ++py)
#pragma unroll
        for (int px = 0; px < 2; ++px) {
            float v = conv2_val(p1, w, bias, b, c, oy * 2 + py, ox * 2 + px);
            v = fmaxf(v * sc + sh, 0.f);
            m = fmaxf(m, v);
        }
    // flatten exactly as reference: h[b, c*49 + oy*7 + ox]
    h[(size_t)b * 1568 + c * 49 + oy * 7 + ox] = m;
}

// ---------------------------------------------------------------------------
// conversions / padding
// ---------------------------------------------------------------------------
__global__ __launch_bounds__(256)
void cvt_f16_kernel(const float* __restrict__ in, _Float16* __restrict__ out, int n) {
    int i = blockIdx.x * 256 + threadIdx.x;
    if (i < n) out[i] = (_Float16)in[i];
}

// zero the K-pad columns [784,800) of feats16 [2048, FPAD]
__global__ __launch_bounds__(256)
void pad_feats16_kernel(_Float16* __restrict__ f) {
    int i = blockIdx.x * 256 + threadIdx.x;
    if (i < BATCH * 16) {
        int b = i / 16, k = 784 + (i % 16);
        f[(size_t)b * FPAD + k] = (_Float16)0.f;
    }
}

// sel_w [10,784] f32 -> padded [16,FPAD] f16 (zeros elsewhere)
__global__ __launch_bounds__(256)
void cvt_selw_kernel(const float* __restrict__ w, _Float16* __restrict__ o) {
    int i = blockIdx.x * 256 + threadIdx.x;
    if (i < 16 * FPAD) {
        int r = i / FPAD, k = i % FPAD;
        o[i] = (r < 10 && k < 784) ? (_Float16)w[r * 784 + k] : (_Float16)0.f;
    }
}

// ---------------------------------------------------------------------------
// fc1: feats16 = f16(tanh(h @ W^T + b)), via v_wmma_f32_16x16x32_f16.
// M=2048, N=784, K=1568. One wave computes a 64x16 tile (4 M-subtiles):
// B (weight) fragment loaded once per K-step, reused by 4 WMMAs.
// Explicit double-buffered fragments so next K-step's loads are in flight
// while the current step's WMMAs execute (avoid s_wait_loadcnt 0).
// ---------------------------------------------------------------------------
__global__ __launch_bounds__(128)
void fc1_wmma_kernel(const _Float16* __restrict__ Ah,   // [2048,1568]
                     const _Float16* __restrict__ Wh,   // [784,1568]
                     const float* __restrict__ bias,    // [784]
                     _Float16* __restrict__ feats16) {  // [2048,FPAD]
    const int lane = threadIdx.x & 31;
    const int wave = threadIdx.x >> 5;
    const int tile = blockIdx.x * 4 + wave;             // 32*49 = 1568 tiles
    const int tM = tile / 49;                           // 64-row block, 0..31
    const int tN = tile % 49;                           // 16-col block, 0..48
    const int m16 = lane & 15;                          // A row / B col / D col
    const int half = lane >> 4;                         // K-half select

    const _Float16* brow  = Wh + (size_t)(tN * 16 + m16) * 1568;
    const _Float16* arow0 = Ah + (size_t)(tM * 64 + m16) * 1568;

    union F { v16h v; v8h h[2]; };
    F fb[2], fa0[2], fa1[2], fa2[2], fa3[2];

    auto load_set = [&](int buf, int k0) {
        fb[buf].h[0] = *(const v8h*)(brow + k0 + half * 8);
        fb[buf].h[1] = *(const v8h*)(brow + k0 + 16 + half * 8);
        const _Float16* p0 = arow0 + k0 + half * 8;
        fa0[buf].h[0] = *(const v8h*)(p0);
        fa0[buf].h[1] = *(const v8h*)(p0 + 16);
        fa1[buf].h[0] = *(const v8h*)(p0 + 16 * 1568);
        fa1[buf].h[1] = *(const v8h*)(p0 + 16 * 1568 + 16);
        fa2[buf].h[0] = *(const v8h*)(p0 + 32 * 1568);
        fa2[buf].h[1] = *(const v8h*)(p0 + 32 * 1568 + 16);
        fa3[buf].h[0] = *(const v8h*)(p0 + 48 * 1568);
        fa3[buf].h[1] = *(const v8h*)(p0 + 48 * 1568 + 16);
    };

    v8f acc0 = {}, acc1 = {}, acc2 = {}, acc3 = {};
    load_set(0, 0);
#pragma unroll 2
    for (int k0 = 0; k0 < 1568; k0 += 32) {
        const int cur = (k0 >> 5) & 1;
        if (k0 + 32 < 1568) load_set(cur ^ 1, k0 + 32);   // prefetch next step
        acc0 = __builtin_amdgcn_wmma_f32_16x16x32_f16(false, fa0[cur].v, false, fb[cur].v,
                                                      (short)0, acc0, false, false);
        acc1 = __builtin_amdgcn_wmma_f32_16x16x32_f16(false, fa1[cur].v, false, fb[cur].v,
                                                      (short)0, acc1, false, false);
        acc2 = __builtin_amdgcn_wmma_f32_16x16x32_f16(false, fa2[cur].v, false, fb[cur].v,
                                                      (short)0, acc2, false, false);
        acc3 = __builtin_amdgcn_wmma_f32_16x16x32_f16(false, fa3[cur].v, false, fb[cur].v,
                                                      (short)0, acc3, false, false);
    }
    const int n = tN * 16 + m16;
    const float bn = bias[n];
    const int mbase = tM * 64 + half * 8;
#pragma unroll
    for (int r = 0; r < 8; ++r) {
        feats16[(size_t)(mbase +  0 + r) * FPAD + n] = (_Float16)tanhf(acc0[r] + bn);
        feats16[(size_t)(mbase + 16 + r) * FPAD + n] = (_Float16)tanhf(acc1[r] + bn);
        feats16[(size_t)(mbase + 32 + r) * FPAD + n] = (_Float16)tanhf(acc2[r] + bn);
        feats16[(size_t)(mbase + 48 + r) * FPAD + n] = (_Float16)tanhf(acc3[r] + bn);
    }
}

// ---------------------------------------------------------------------------
// sel = feats @ sel_w^T + sel_b via WMMA: M=2048, N=16(10 valid), K=FPAD=800
// ---------------------------------------------------------------------------
__global__ __launch_bounds__(128)
void sel_wmma_kernel(const _Float16* __restrict__ F,    // [2048,FPAD]
                     const _Float16* __restrict__ Wp,   // [16,FPAD] padded
                     const float* __restrict__ sb,      // [10]
                     float* __restrict__ sel) {         // [2048,10]
    const int lane = threadIdx.x & 31;
    const int wave = threadIdx.x >> 5;
    const int tM = blockIdx.x * 4 + wave;               // 128 M-tiles
    const int m16 = lane & 15;
    const int half = lane >> 4;

    const _Float16* arow = F + (size_t)(tM * 16 + m16) * FPAD;
    const _Float16* brow = Wp + (size_t)m16 * FPAD;

    union Fu { v16h v; v8h h[2]; };
    Fu a[2], b[2];
    auto load_set = [&](int buf, int k0) {
        a[buf].h[0] = *(const v8h*)(arow + k0 + half * 8);
        a[buf].h[1] = *(const v8h*)(arow + k0 + 16 + half * 8);
        b[buf].h[0] = *(const v8h*)(brow + k0 + half * 8);
        b[buf].h[1] = *(const v8h*)(brow + k0 + 16 + half * 8);
    };

    v8f acc = {};
    load_set(0, 0);
#pragma unroll 2
    for (int k0 = 0; k0 < FPAD; k0 += 32) {
        const int cur = (k0 >> 5) & 1;
        if (k0 + 32 < FPAD) load_set(cur ^ 1, k0 + 32);
        acc = __builtin_amdgcn_wmma_f32_16x16x32_f16(false, a[cur].v, false, b[cur].v,
                                                     (short)0, acc, false, false);
    }
    const int n = m16;
    if (n < 10) {
        const float bn = sb[n];
#pragma unroll
        for (int r = 0; r < 8; ++r) {
            int m = tM * 16 + r + half * 8;
            sel[(size_t)m * 10 + n] = acc[r] + bn;
        }
    }
}

// ---------------------------------------------------------------------------
// 10-qubit real-statevector simulator: one block (256 thr) per sample.
// axis i of (2,)*10 state  <->  bit (9-i) of the flat index.
// ---------------------------------------------------------------------------
__device__ __forceinline__ void apply_ry(float* st, float theta, int wire, int tid) {
    int p = 9 - wire;
    float c, s;
    __sincosf(theta * 0.5f, &s, &c);
    for (int t = tid; t < 512; t += 256) {
        int low = t & ((1 << p) - 1);
        int hi  = t >> p;
        int i0 = (hi << (p + 1)) | low;
        int i1 = i0 | (1 << p);
        float a0 = st[i0], a1 = st[i1];
        st[i0] = c * a0 - s * a1;
        st[i1] = s * a0 + c * a1;
    }
    __syncthreads();
}

__global__ __launch_bounds__(256)
void quantum_kernel(const float* __restrict__ sel, const float* __restrict__ qp,
                    float* __restrict__ qout) {
    __shared__ float st[1024];
    __shared__ float red[256][10];
    int b = blockIdx.x, tid = threadIdx.x;
    for (int i = tid; i < 1024; i += 256) st[i] = 0.f;
    __syncthreads();
    if (tid == 0) st[0] = 1.f;
    __syncthreads();
    for (int q = 0; q < 10; ++q)
        apply_ry(st, sel[(size_t)b * 10 + q] * PI_F, q, tid);
    for (int L = 0; L < 3; ++L) {
        for (int q = 0; q < 10; ++q)
            apply_ry(st, qp[L * 10 + q], q, tid);
        // CZ(i,i+1) chain fused: sign flip if adjacent-1-pair count is odd
        for (int i = tid; i < 1024; i += 256) {
            int par = __popc(i & (i >> 1) & 0x1FF) & 1;
            if (par) st[i] = -st[i];
        }
        __syncthreads();
    }
    float acc[10];
#pragma unroll
    for (int q = 0; q < 10; ++q) acc[q] = 0.f;
    for (int i = tid; i < 1024; i += 256) {
        float pr = st[i] * st[i];
#pragma unroll
        for (int q = 0; q < 10; ++q)
            acc[q] += ((i >> (9 - q)) & 1) ? -pr : pr;
    }
#pragma unroll
    for (int q = 0; q < 10; ++q) red[tid][q] = acc[q];
    __syncthreads();
    if (tid < 10) {
        float s = 0.f;
        for (int j = 0; j < 256; ++j) s += red[j][tid];
        qout[(size_t)b * 10 + tid] = s;
    }
}

// ---------------------------------------------------------------------------
// post: out = relu(q @ p1w^T + p1b) @ p2w^T + p2b
// ---------------------------------------------------------------------------
__global__ __launch_bounds__(256)
void post_kernel(const float* __restrict__ qout, const float* __restrict__ w1,
                 const float* __restrict__ b1, const float* __restrict__ w2,
                 const float* __restrict__ b2, float* __restrict__ out) {
    int b = blockIdx.x * 256 + threadIdx.x;
    if (b >= BATCH) return;
    float q[10];
#pragma unroll
    for (int i = 0; i < 10; ++i) q[i] = qout[(size_t)b * 10 + i];
    float h2[32];
#pragma unroll
    for (int i = 0; i < 32; ++i) {
        float s = b1[i];
#pragma unroll
        for (int k = 0; k < 10; ++k) s += w1[i * 10 + k] * q[k];
        h2[i] = fmaxf(s, 0.f);
    }
#pragma unroll
    for (int j = 0; j < 10; ++j) {
        float s = b2[j];
#pragma unroll
        for (int i = 0; i < 32; ++i) s += w2[j * 32 + i] * h2[i];
        out[(size_t)b * 10 + j] = s;
    }
}

// ---------------------------------------------------------------------------
// launcher
// ---------------------------------------------------------------------------
extern "C" void kernel_launch(void* const* d_in, const int* in_sizes, int n_in,
                              void* d_out, int out_size, void* d_ws, size_t ws_size,
                              hipStream_t stream) {
    const float* x       = (const float*)d_in[0];
    const float* c1w     = (const float*)d_in[1];
    const float* c1b     = (const float*)d_in[2];
    const float* bn1g    = (const float*)d_in[3];
    const float* bn1b    = (const float*)d_in[4];
    const float* c2w     = (const float*)d_in[5];
    const float* c2b     = (const float*)d_in[6];
    const float* bn2g    = (const float*)d_in[7];
    const float* bn2b    = (const float*)d_in[8];
    const float* fc1w    = (const float*)d_in[9];
    const float* fc1b    = (const float*)d_in[10];
    const float* selw    = (const float*)d_in[11];
    const float* selb    = (const float*)d_in[12];
    const float* qparams = (const float*)d_in[13];
    const float* p1w     = (const float*)d_in[14];
    const float* p1b     = (const float*)d_in[15];
    const float* p2w     = (const float*)d_in[16];
    const float* p2b     = (const float*)d_in[17];
    float* out = (float*)d_out;

    // workspace layout (bytes), ~48 MB peak
    char* ws = (char*)d_ws;
    size_t o = 0;
    float*    pool1   = (float*)(ws + o);    o += (size_t)BATCH * 16 * 14 * 14 * 4; // 25.7 MB
    float*    hflat   = (float*)(ws + o);    o += (size_t)BATCH * 1568 * 4;         // 12.8 MB
    _Float16* hf16    = (_Float16*)(ws + o); o += (size_t)BATCH * 1568 * 2;         // 6.4 MB
    _Float16* wf16    = (_Float16*)(ws + o); o += (size_t)784 * 1568 * 2;           // 2.4 MB
    _Float16* feats16 = (_Float16*)(ws + o); o += (size_t)BATCH * FPAD * 2;         // 3.3 MB
    _Float16* selw16  = (_Float16*)(ws + o); o += (size_t)16 * FPAD * 2;
    float*    sel     = (float*)(ws + o);    o += (size_t)BATCH * 10 * 4;
    float*    qout    = (float*)(ws + o);    o += (size_t)BATCH * 10 * 4;
    float*    part1   = (float*)(ws + o);    o += (size_t)16 * 128 * 2 * 4;
    float*    part2   = (float*)(ws + o);    o += (size_t)32 * 128 * 2 * 4;
    float*    scale1  = (float*)(ws + o);    o += 16 * 4;
    float*    shift1  = (float*)(ws + o);    o += 16 * 4;
    float*    scale2  = (float*)(ws + o);    o += 32 * 4;
    float*    shift2  = (float*)(ws + o);    o += 32 * 4;

    // conv1: batch stats -> bn scale/shift -> fused conv+bn+relu+pool
    conv1_stats_kernel<<<16 * 128, 256, 0, stream>>>(x, c1w, c1b, part1);
    bn_finalize_kernel<<<16, 128, 0, stream>>>(part1, bn1g, bn1b, scale1, shift1,
                                               1.f / (float)(BATCH * 784));
    pool1_kernel<<<(BATCH * 16 * 14 * 14) / 256, 256, 0, stream>>>(x, c1w, c1b,
                                                                   scale1, shift1, pool1);
    // conv2: same scheme -> flattened h [2048,1568]
    conv2_stats_kernel<<<32 * 128, 256, 0, stream>>>(pool1, c2w, c2b, part2);
    bn_finalize_kernel<<<32, 128, 0, stream>>>(part2, bn2g, bn2b, scale2, shift2,
                                               1.f / (float)(BATCH * 196));
    pool2_kernel<<<(BATCH * 32 * 7 * 7) / 256, 256, 0, stream>>>(pool1, c2w, c2b,
                                                                 scale2, shift2, hflat);
    // f16 operands for WMMA
    cvt_f16_kernel<<<(BATCH * 1568 + 255) / 256, 256, 0, stream>>>(hflat, hf16,
                                                                   BATCH * 1568);
    cvt_f16_kernel<<<(784 * 1568 + 255) / 256, 256, 0, stream>>>(fc1w, wf16,
                                                                 784 * 1568);
    pad_feats16_kernel<<<(BATCH * 16 + 255) / 256, 256, 0, stream>>>(feats16);
    cvt_selw_kernel<<<(16 * FPAD + 255) / 256, 256, 0, stream>>>(selw, selw16);
    // fc1 GEMM (WMMA, 64x16 per wave, double-buffered) + fused bias+tanh
    fc1_wmma_kernel<<<(32 * 49) / 4, 128, 0, stream>>>(hf16, wf16, fc1b, feats16);
    // sel GEMM (WMMA, double-buffered) -> [2048,10]
    sel_wmma_kernel<<<128 / 4, 128, 0, stream>>>(feats16, selw16, selb, sel);
    // 10-qubit statevector simulation, one block per sample
    quantum_kernel<<<BATCH, 256, 0, stream>>>(sel, qparams, qout);
    // post MLP -> d_out [2048,10]
    post_kernel<<<(BATCH + 255) / 256, 256, 0, stream>>>(qout, p1w, p1b, p2w, p2b, out);
}
